// GCN_27754078667427
// MI455X (gfx1250) — compile-verified, matrix operand order checked by reading, not run on hardware
//
#include <hip/hip_runtime.h>
#include <hip/hip_bf16.h>

#define NN 100000
#define EE 1600000
#define DIN 128
#define DH  256
#define GG  512

typedef __attribute__((ext_vector_type(16))) __bf16    v16bf;
typedef __attribute__((ext_vector_type(8)))  float     v8f;
typedef __attribute__((ext_vector_type(8)))  unsigned short ushort8;

union FragBF { v16bf bf; ushort8 h[2]; };

__device__ __forceinline__ unsigned short f32_to_bf16(float f) {
    unsigned int u = __float_as_uint(f);
    // round-to-nearest-even
    unsigned int r = (u + 0x7FFFu + ((u >> 16) & 1u)) >> 16;
    return (unsigned short)r;
}

// Guaranteed native CDNA5 f32 atomic add (no-return form, GV addressing).
// Avoids any chance of a CAS-loop lowering on the bandwidth-critical scatter path.
__device__ __forceinline__ void atomic_add_f32(float* p, float v) {
    asm volatile("global_atomic_add_f32 %0, %1, off"
                 :
                 : "v"(p), "v"(v)
                 : "memory");
}

// ---------- weight prep: W [K x 256] f32 row-major -> Wt [256 x K] bf16 ----------
__global__ void k_transpose_w(const float* __restrict__ W, unsigned short* __restrict__ Wt, int K) {
    int tid = blockIdx.x * blockDim.x + threadIdx.x;
    if (tid >= K * DH) return;
    int col = tid / K;      // output column (0..255)
    int k   = tid % K;      // input dim
    Wt[(size_t)col * K + k] = f32_to_bf16(W[(size_t)k * DH + col]);
}

// ---------- degree / normalization ----------
__global__ void k_deg_init(float* __restrict__ deg) {
    int n = blockIdx.x * blockDim.x + threadIdx.x;
    if (n < NN) deg[n] = 1.0f;   // self-loop
}
__global__ void k_deg_count(const int* __restrict__ dst, float* __restrict__ deg) {
    int e = blockIdx.x * blockDim.x + threadIdx.x;
    if (e < EE) atomic_add_f32(&deg[dst[e]], 1.0f);
}
__global__ void k_deg_fin(float* __restrict__ deg) {
    int n = blockIdx.x * blockDim.x + threadIdx.x;
    if (n < NN) deg[n] = rsqrtf(deg[n]);   // now holds d^{-1/2}
}

// ---------- WMMA GEMM: C[N x 256] = A[N x K] (f32, cvt->bf16) @ W (bf16, pre-transposed) ----------
// grid.x = N/16 row tiles; block = 128 threads = 4 waves; wave w covers cols [w*64, w*64+64)
__global__ void k_gemm_bf16(const float* __restrict__ A,
                            const unsigned short* __restrict__ Wt,
                            float* __restrict__ C, int K) {
    __shared__ unsigned short lds_a[16 * 32];   // 16 rows x 32 k, bf16
    const int lane = threadIdx.x & 31;
    const int wave = threadIdx.x >> 5;
    const int rowbase = blockIdx.x * 16;
    const int colbase = wave * 64;

    v8f acc[4] = {};

    const int row = lane & 15;
    const int kk0 = (lane >> 4) << 3;           // 0 or 8 (K-half split per CDNA5 A layout)

    for (int kb = 0; kb < K; kb += 32) {
        __syncthreads();
        // cooperative stage of A tile (16x32 f32) into LDS as bf16
        {
            int idx = threadIdx.x * 4;          // 512 elems / 128 threads
            int r = idx >> 5, kk = idx & 31;
            const float4 v = *(const float4*)(A + (size_t)(rowbase + r) * K + kb + kk);
            unsigned short* d = &lds_a[r * 32 + kk];
            d[0] = f32_to_bf16(v.x); d[1] = f32_to_bf16(v.y);
            d[2] = f32_to_bf16(v.z); d[3] = f32_to_bf16(v.w);
        }
        __syncthreads();

        FragBF a;
        a.h[0] = *(const ushort8*)&lds_a[row * 32 + kk0];
        a.h[1] = *(const ushort8*)&lds_a[row * 32 + kk0 + 16];

        #pragma unroll
        for (int c = 0; c < 4; ++c) {
            const int colN = colbase + c * 16 + (lane & 15);
            const unsigned short* bp = Wt + (size_t)colN * K + kb + kk0;
            FragBF b;
            b.h[0] = *(const ushort8*)bp;
            b.h[1] = *(const ushort8*)(bp + 16);
            acc[c] = __builtin_amdgcn_wmma_f32_16x16x32_bf16(
                false, a.bf, false, b.bf, (short)0, acc[c], false, false);
        }
    }

    // C/D layout: VGPR r -> (M = r + 8*(lane>>4), N = lane&15) within 16x16 tile
    const int mofs = (lane >> 4) << 3;
    #pragma unroll
    for (int c = 0; c < 4; ++c) {
        #pragma unroll
        for (int r = 0; r < 8; ++r) {
            C[(size_t)(rowbase + r + mofs) * DH + colbase + c * 16 + (lane & 15)] = acc[c][r];
        }
    }
}

// ---------- aggregation: self-loop term initializes output (also zeroes stale ws) ----------
__global__ void k_selfloop(const float* __restrict__ dinv, const float* __restrict__ h,
                           float* __restrict__ out) {
    int n = blockIdx.x, d = threadIdx.x;
    float s = dinv[n];
    out[(size_t)n * DH + d] = s * s * h[(size_t)n * DH + d];
}

// one wave per edge: gather h[src] (2x float4/lane), scatter-add into out[dst]
__global__ void k_edge_scatter(const int* __restrict__ src, const int* __restrict__ dst,
                               const float* __restrict__ dinv, const float* __restrict__ h,
                               float* __restrict__ out) {
    int e = blockIdx.x * (blockDim.x >> 5) + (threadIdx.x >> 5);
    if (e >= EE) return;
    int lane = threadIdx.x & 31;
    int s = src[e], d = dst[e];
    float nrm = dinv[s] * dinv[d];
    const float4* hs = (const float4*)(h + (size_t)s * DH);
    float* od = out + (size_t)d * DH;
    #pragma unroll
    for (int i = 0; i < 2; ++i) {
        int q = lane + i * 32;              // 64 float4 per row
        float4 v = hs[q];
        int base = q * 4;
        atomic_add_f32(od + base + 0, nrm * v.x);
        atomic_add_f32(od + base + 1, nrm * v.y);
        atomic_add_f32(od + base + 2, nrm * v.z);
        atomic_add_f32(od + base + 3, nrm * v.w);
    }
}

__global__ void k_bias_relu(float* __restrict__ out, const float* __restrict__ b) {
    int n = blockIdx.x, d = threadIdx.x;
    float v = out[(size_t)n * DH + d] + b[d];
    out[(size_t)n * DH + d] = v > 0.0f ? v : 0.0f;
}

// ---------- pooling + head ----------
__global__ void k_pool_zero(float* __restrict__ sums, float* __restrict__ cnts) {
    int g = blockIdx.x, d = threadIdx.x;
    sums[(size_t)g * DH + d] = 0.0f;
    if (d == 0) cnts[g] = 0.0f;
}
__global__ void k_pool_acc(const int* __restrict__ batch, const float* __restrict__ h,
                           float* __restrict__ sums, float* __restrict__ cnts) {
    int n = blockIdx.x, d = threadIdx.x;
    int g = batch[n];
    atomic_add_f32(&sums[(size_t)g * DH + d], h[(size_t)n * DH + d]);
    if (d == 0) atomic_add_f32(&cnts[g], 1.0f);
}
__global__ void k_head(const float* __restrict__ sums, const float* __restrict__ cnts,
                       const float* __restrict__ Wl, const float* __restrict__ bl,
                       float* __restrict__ out) {
    int g = blockIdx.x, lane = threadIdx.x;   // 32 threads
    float acc = 0.0f;
    #pragma unroll
    for (int j = 0; j < 8; ++j) {
        int d = lane + j * 32;
        acc += sums[(size_t)g * DH + d] * Wl[d];
    }
    #pragma unroll
    for (int off = 16; off > 0; off >>= 1)
        acc += __shfl_down(acc, off, 32);
    if (lane == 0) {
        float c = cnts[g];
        if (c < 1.0f) c = 1.0f;
        out[g] = acc / c + bl[0];
    }
}

extern "C" void kernel_launch(void* const* d_in, const int* in_sizes, int n_in,
                              void* d_out, int out_size, void* d_ws, size_t ws_size,
                              hipStream_t stream) {
    const float* x   = (const float*)d_in[0];
    const int*   ei  = (const int*)d_in[1];      // [2, E] flat: [0..E)=src, [E..2E)=dst
    const int*   bat = (const int*)d_in[2];
    const float* W1  = (const float*)d_in[3];
    const float* b1  = (const float*)d_in[4];
    const float* W2  = (const float*)d_in[5];
    const float* b2  = (const float*)d_in[6];
    const float* Wl  = (const float*)d_in[7];
    const float* bl  = (const float*)d_in[8];
    float* out = (float*)d_out;

    const int* src = ei;
    const int* dst = ei + EE;

    // workspace carve-up (256B aligned)
    char* w = (char*)d_ws;
    auto alloc = [&](size_t bytes) {
        void* p = (void*)w;
        w += (bytes + 255) & ~(size_t)255;
        return p;
    };
    float*          dinv = (float*)alloc((size_t)NN * 4);            // deg -> d^-1/2
    float*          bufA = (float*)alloc((size_t)NN * DH * 4);       // GEMM outputs
    float*          bufB = (float*)alloc((size_t)NN * DH * 4);       // aggregated activations
    unsigned short* w1t  = (unsigned short*)alloc((size_t)DH * DIN * 2);
    unsigned short* w2t  = (unsigned short*)alloc((size_t)DH * DH * 2);
    float*          sums = (float*)alloc((size_t)GG * DH * 4);
    float*          cnts = (float*)alloc((size_t)GG * 4);

    // weight prep
    k_transpose_w<<<(DIN * DH + 255) / 256, 256, 0, stream>>>(W1, w1t, DIN);
    k_transpose_w<<<(DH * DH + 255) / 256, 256, 0, stream>>>(W2, w2t, DH);

    // degree normalization
    k_deg_init<<<(NN + 255) / 256, 256, 0, stream>>>(dinv);
    k_deg_count<<<(EE + 255) / 256, 256, 0, stream>>>(dst, dinv);
    k_deg_fin<<<(NN + 255) / 256, 256, 0, stream>>>(dinv);

    // layer 1: h1 = x @ W1 ; aggregate ; +b1 ; relu
    k_gemm_bf16<<<NN / 16, 128, 0, stream>>>(x, w1t, bufA, DIN);
    k_selfloop<<<NN, DH, 0, stream>>>(dinv, bufA, bufB);
    k_edge_scatter<<<(EE + 7) / 8, 256, 0, stream>>>(src, dst, dinv, bufA, bufB);
    k_bias_relu<<<NN, DH, 0, stream>>>(bufB, b1);

    // layer 2: h2 = h1 @ W2 ; aggregate ; +b2 ; relu
    k_gemm_bf16<<<NN / 16, 128, 0, stream>>>(bufB, w2t, bufA, DH);
    k_selfloop<<<NN, DH, 0, stream>>>(dinv, bufA, bufB);
    k_edge_scatter<<<(EE + 7) / 8, 256, 0, stream>>>(src, dst, dinv, bufA, bufB);
    k_bias_relu<<<NN, DH, 0, stream>>>(bufB, b2);

    // mean pool + head
    k_pool_zero<<<GG, DH, 0, stream>>>(sums, cnts);
    k_pool_acc<<<NN, DH, 0, stream>>>(bat, bufB, sums, cnts);
    k_head<<<GG, 32, 0, stream>>>(sums, cnts, Wl, bl, out);
}
